// MPNN_66211215835311
// MI455X (gfx1250) — matrix operand order repaired and checked by reading, not compiled, not run on hardware
//
#include <hip/hip_runtime.h>
#include <math.h>

// MPNN (REANN-style). Edge passes are L2-atomic bound -> recompute geometry each
// pass (~12MB workspace, all hot state L2-resident); dense per-atom MLPs use
// f16 WMMA (v_wmma_f32_16x16x32_f16) with LDS-staged activations + transposed
// weights so both A and B fragments are contiguous ds_load_b128s.

#define NEDGE_BLK 256

typedef __attribute__((ext_vector_type(16))) _Float16 v16h;
typedef __attribute__((ext_vector_type(8)))  float    v8f;

__device__ __forceinline__ float silu_f(float x) { return x / (1.f + __expf(-x)); }

__device__ __forceinline__ void ln_silu8(float* h) {
  float mu = 0.f;
#pragma unroll
  for (int j = 0; j < 8; ++j) mu += h[j];
  mu *= 0.125f;
  float var = 0.f;
#pragma unroll
  for (int j = 0; j < 8; ++j) { float d = h[j] - mu; var += d * d; }
  var *= 0.125f;
  float r = rsqrtf(var + 1e-5f);
#pragma unroll
  for (int j = 0; j < 8; ++j) h[j] = silu_f((h[j] - mu) * r);
}

// ---------------- per-atom embedding MLPs (tiny: 1->8->8->{24,64}) -------------
__global__ void embed_kernel(const float* __restrict__ species,
    const float* __restrict__ Wn0, const float* __restrict__ bn0,
    const float* __restrict__ Wn1, const float* __restrict__ bn1,
    const float* __restrict__ Wn2, const float* __restrict__ bn2,
    const float* __restrict__ Wc0, const float* __restrict__ bc0,
    const float* __restrict__ Wc1, const float* __restrict__ bc1,
    const float* __restrict__ Wc2, const float* __restrict__ bc2,
    float* __restrict__ ncoef, float* __restrict__ ccoef, int natoms)
{
  int a = blockIdx.x * blockDim.x + threadIdx.x;
  if (a >= natoms) return;
  float s = species[a];
  float h[8], g[8];
  // neigh: 1->8->8->24
#pragma unroll
  for (int j = 0; j < 8; ++j) h[j] = s * Wn0[j] + bn0[j];
  ln_silu8(h);
#pragma unroll
  for (int j = 0; j < 8; ++j) {
    float acc = bn1[j];
#pragma unroll
    for (int i = 0; i < 8; ++i) acc += h[i] * Wn1[i * 8 + j];
    g[j] = acc;
  }
  ln_silu8(g);
  for (int m = 0; m < 24; ++m) {
    float acc = bn2[m];
#pragma unroll
    for (int i = 0; i < 8; ++i) acc += g[i] * Wn2[i * 24 + m];
    ncoef[a * 24 + m] = acc;
  }
  // center: 1->8->8->64
#pragma unroll
  for (int j = 0; j < 8; ++j) h[j] = s * Wc0[j] + bc0[j];
  ln_silu8(h);
#pragma unroll
  for (int j = 0; j < 8; ++j) {
    float acc = bc1[j];
#pragma unroll
    for (int i = 0; i < 8; ++i) acc += h[i] * Wc1[i * 8 + j];
    g[j] = acc;
  }
  ln_silu8(g);
  for (int m = 0; m < 64; ++m) {
    float acc = bc2[m];
#pragma unroll
    for (int i = 0; i < 8; ++i) acc += g[i] * Wc2[i * 64 + m];
    ccoef[a * 64 + m] = acc;
  }
}

// ---------------- edge geometry (recomputed each pass; VALU is free) -----------
__device__ __forceinline__ void edge_geom(int e, const float* __restrict__ cart,
    const int* __restrict__ nl, const float* __restrict__ shifts,
    const float* __restrict__ neigh_factor, const float* __restrict__ ncoef,
    int nedges, int& ic, int& in_, float* crj, float* sph, float& cut)
{
  ic = nl[e];
  in_ = nl[nedges + e];
  float dx = cart[in_ * 3 + 0] - cart[ic * 3 + 0] + shifts[e * 3 + 0];
  float dy = cart[in_ * 3 + 1] - cart[ic * 3 + 1] + shifts[e * 3 + 1];
  float dz = cart[in_ * 3 + 2] - cart[ic * 3 + 2] + shifts[e * 3 + 2];
  float d = sqrtf(dx * dx + dy * dy + dz * dz);
  float inv = 1.f / (d + 1e-9f);
  float x = dx * inv, y = dy * inv, z = dz * inv;
  float tc = 0.5f * __cosf(d * 0.78539816339f) + 0.5f; // pi/CUTOFF, CUTOFF=4
  cut = neigh_factor[e] * tc * tc;
  const float* ne = ncoef + (size_t)in_ * 24;
#pragma unroll
  for (int j = 0; j < 8; ++j) {
    float al = ne[8 + j], rs = ne[16 + j];
    float tt = al * (d - rs);
    crj[j] = cut * __expf(-tt * tt) * ne[j];   // cut * radial * emb_w
  }
  sph[0] = 0.28209479177f;
  sph[1] = 0.48860251190f * y;
  sph[2] = 0.48860251190f * z;
  sph[3] = 0.48860251190f * x;
  sph[4] = 1.09254843059f * x * y;
  sph[5] = 1.09254843059f * y * z;
  sph[6] = 0.31539156525f * (3.f * z * z - 1.f);
  sph[7] = 1.09254843059f * x * z;
  sph[8] = 0.54627421529f * (x * x - y * y);
}

__global__ void edge_scatter_kernel(const float* __restrict__ cart,
    const int* __restrict__ nl, const float* __restrict__ shifts,
    const float* __restrict__ neigh_factor, const float* __restrict__ ncoef,
    float* __restrict__ co, int nedges)
{
  int e = blockIdx.x * blockDim.x + threadIdx.x;
  if (e >= nedges) return;
  int ic, in_;
  float crj[8], sph[9], cut;
  edge_geom(e, cart, nl, shifts, neigh_factor, ncoef, nedges, ic, in_, crj, sph, cut);
  float* dst = co + (size_t)ic * 72;
#pragma unroll
  for (int k = 0; k < 9; ++k) {
    float s = sph[k];
#pragma unroll
    for (int j = 0; j < 8; ++j)
      unsafeAtomicAdd(&dst[k * 8 + j], crj[j] * s);
  }
}

__global__ void iter_edge_kernel(const float* __restrict__ cart,
    const int* __restrict__ nl, const float* __restrict__ shifts,
    const float* __restrict__ neigh_factor, const float* __restrict__ ncoef,
    const float* __restrict__ icoef, const float* __restrict__ co_old,
    float* __restrict__ co_new, int nedges)
{
  int e = blockIdx.x * blockDim.x + threadIdx.x;
  if (e >= nedges) return;
  int ic, in_;
  float crj[8], sph[9], cut;
  edge_geom(e, cart, nl, shifts, neigh_factor, ncoef, nedges, ic, in_, crj, sph, cut);
  const float* icf = icoef + (size_t)in_ * 8;
  const float* old = co_old + (size_t)in_ * 72;
  float* dst = co_new + (size_t)ic * 72;
  float wj[8];
#pragma unroll
  for (int j = 0; j < 8; ++j) wj[j] = icf[j] * crj[j];
#pragma unroll
  for (int k = 0; k < 9; ++k) {
    float s = sph[k];
#pragma unroll
    for (int j = 0; j < 8; ++j)
      unsafeAtomicAdd(&dst[k * 8 + j], wj[j] * s + old[k * 8 + j] * cut);
  }
}

// ------- contraction + density: density[i,m] += ccoef[i,m] * sum_k (co@cc)^2 ---
__global__ void contract_density_kernel(const float* __restrict__ co,
    const float* __restrict__ cc /* (3,8,64) slice */,
    const float* __restrict__ ccoef, float* __restrict__ density, int natoms)
{
  int t = blockIdx.x * blockDim.x + threadIdx.x;
  if (t >= natoms * 64) return;
  int i = t >> 6, m = t & 63;
  const float* row = co + (size_t)i * 72;
  float acc = 0.f;
#pragma unroll
  for (int k = 0; k < 9; ++k) {
    int l = (k == 0) ? 0 : ((k < 4) ? 1 : 2);           // INDEX_L
    const float* w = cc + (size_t)(l * 8) * 64 + m;     // cc[l][j][m]
    float s = 0.f;
#pragma unroll
    for (int j = 0; j < 8; ++j) s += row[k * 8 + j] * w[j * 64];
    acc += s * s;
  }
  density[(size_t)i * 64 + m] += acc * ccoef[(size_t)i * 64 + m];
}

// ------------- dense MLP 64->64->64->64->{8|1} over atoms via WMMA f16 --------
// 128 threads = 4 waves; each wave owns 16 atoms; per layer: 2 K-chunks x 4 N-tiles
// of v_wmma_f32_16x16x32_f16.  Weights staged TRANSPOSED (n-major) in LDS so each
// B fragment is one contiguous 32B v16h load; A fragments are contiguous too.
// LayerNorm+SiLU in f32.  EXEC stays all-ones (no early exits before WMMA).
__global__ __launch_bounds__(128) void mlp_wmma_kernel(
    const float* __restrict__ X,
    const float* __restrict__ W0, const float* __restrict__ B0,
    const float* __restrict__ W1, const float* __restrict__ B1,
    const float* __restrict__ W2, const float* __restrict__ B2,
    const float* __restrict__ W3, const float* __restrict__ B3,
    int outw, float* __restrict__ out,
    const float* __restrict__ center_factor, int natoms)
{
  __shared__ __align__(32) _Float16 sWt[64 * 64];   // sWt[n*64 + k] = W[k][n]
  __shared__ __align__(32) _Float16 sAct[64 * 64];  // sAct[atom*64 + k]
  __shared__ float    sOut[64 * 64];
  __shared__ float    sBias[64];

  const int tid  = threadIdx.x;
  const int lane = tid & 31;
  const int wave = tid >> 5;
  const int atomBase = blockIdx.x * 64;
  const int aBase = wave * 16;
  const int m    = lane & 15;
  const int half = lane >> 4;

  // stage input activations (f32 -> f16), zero-pad past natoms
  for (int i = tid; i < 64 * 64; i += 128) {
    int al = i >> 6, k = i & 63;
    int atom = atomBase + al;
    float v = (atom < natoms) ? X[(size_t)atom * 64 + k] : 0.f;
    sAct[i] = (_Float16)v;
  }
  __syncthreads();

  const float* Ws[3] = { W0, W1, W2 };
  const float* Bs[3] = { B0, B1, B2 };

#pragma unroll
  for (int L = 0; L < 3; ++L) {
    // coalesced global read (k-major), transposed LDS write (n-major)
    for (int i = tid; i < 64 * 64; i += 128) {
      int k = i >> 6, n = i & 63;
      sWt[n * 64 + k] = (_Float16)Ws[L][i];
    }
    if (tid < 64) sBias[tid] = Bs[L][tid];
    __syncthreads();

    v8f acc[4] = {};
#pragma unroll
    for (int kc = 0; kc < 2; ++kc) {
      // A fragment: 16 atoms x K32 (contiguous halves: k = 16*half + s)
      const _Float16* row = &sAct[(aBase + m) * 64 + kc * 32 + half * 16];
      v16h afrag;
#pragma unroll
      for (int r = 0; r < 4; ++r) {
        afrag[2 * r]     = row[2 * r];
        afrag[2 * r + 1] = row[2 * r + 1];
        afrag[8 + 2 * r]     = row[8 + 2 * r];
        afrag[8 + 2 * r + 1] = row[8 + 2 * r + 1];
      }
      const int kb = kc * 32 + half * 16;
#pragma unroll
      for (int nt = 0; nt < 4; ++nt) {
        const int n = m + 16 * nt;
        // B fragment: 16 contiguous halves of the transposed weight row
        v16h bfrag = *reinterpret_cast<const v16h*>(&sWt[n * 64 + kb]);
        acc[nt] = __builtin_amdgcn_wmma_f32_16x16x32_f16(
            false, afrag, false, bfrag, (short)0, acc[nt], false, false);
      }
    }
    // D layout: vgpr r, lane -> M = r + 8*half, N = (lane&15) + 16*nt
#pragma unroll
    for (int nt = 0; nt < 4; ++nt)
#pragma unroll
      for (int r = 0; r < 8; ++r)
        sOut[(aBase + r + 8 * half) * 64 + m + 16 * nt] = acc[nt][r];
    __syncthreads();

    // bias + LayerNorm + SiLU (f32), back to f16 activations
    if (tid < 64) {
      float mu = 0.f, sq = 0.f;
      for (int n = 0; n < 64; ++n) {
        float v = sOut[tid * 64 + n] + sBias[n];
        mu += v; sq += v * v;
      }
      mu *= (1.f / 64.f);
      float var = sq * (1.f / 64.f) - mu * mu;
      float rr = rsqrtf(var + 1e-5f);
      for (int n = 0; n < 64; ++n) {
        float v = sOut[tid * 64 + n] + sBias[n];
        sAct[tid * 64 + n] = (_Float16)silu_f((v - mu) * rr);
      }
    }
    __syncthreads();
  }

  // final thin layer (outw = 8 -> iter_coeff, outw = 1 -> reduce into d_out)
  if (outw == 8) {
    for (int idx = tid; idx < 64 * 8; idx += 128) {
      int al = idx >> 3, j = idx & 7;
      int atom = atomBase + al;
      if (atom < natoms) {
        float v = B3[j];
#pragma unroll 8
        for (int k = 0; k < 64; ++k) v += (float)sAct[al * 64 + k] * W3[k * 8 + j];
        out[(size_t)atom * 8 + j] = v;
      }
    }
  } else {
    if (tid < 64) {
      int atom = atomBase + tid;
      if (atom < natoms) {
        float v = B3[0];
#pragma unroll 8
        for (int k = 0; k < 64; ++k) v += (float)sAct[tid * 64 + k] * W3[k];
        unsafeAtomicAdd(out, v * center_factor[atom]);
      }
    }
  }
}

// ------------------------------------------------------------------------------
extern "C" void kernel_launch(void* const* d_in, const int* in_sizes, int n_in,
                              void* d_out, int out_size, void* d_ws, size_t ws_size,
                              hipStream_t stream) {
  const float* cart          = (const float*)d_in[0];
  const int*   nl            = (const int*)  d_in[1];
  const float* shifts        = (const float*)d_in[2];
  const float* center_factor = (const float*)d_in[3];
  const float* neigh_factor  = (const float*)d_in[4];
  const float* species       = (const float*)d_in[5];
  const float* cc            = (const float*)d_in[6];   // (4,3,8,64)
  // emb_neigh params: d_in[7..12], emb_center: d_in[13..18]
  // iter_params[t]: d_in[19+8t .. 26+8t], out_params: d_in[43..50]

  const int natoms = in_sizes[0] / 3;
  const int nedges = in_sizes[4];

  float* ws      = (float*)d_ws;
  float* co_a    = ws;                                   // natoms*72
  float* co_b    = co_a    + (size_t)natoms * 72;        // natoms*72
  float* density = co_b    + (size_t)natoms * 72;        // natoms*64
  float* ccoef   = density + (size_t)natoms * 64;        // natoms*64
  float* ncoef   = ccoef   + (size_t)natoms * 64;        // natoms*24
  float* icoef   = ncoef   + (size_t)natoms * 24;        // natoms*8

  hipMemsetAsync(co_a,    0, (size_t)natoms * 72 * sizeof(float), stream);
  hipMemsetAsync(density, 0, (size_t)natoms * 64 * sizeof(float), stream);

  embed_kernel<<<(natoms + 255) / 256, 256, 0, stream>>>(species,
      (const float*)d_in[7],  (const float*)d_in[8],  (const float*)d_in[9],
      (const float*)d_in[10], (const float*)d_in[11], (const float*)d_in[12],
      (const float*)d_in[13], (const float*)d_in[14], (const float*)d_in[15],
      (const float*)d_in[16], (const float*)d_in[17], (const float*)d_in[18],
      ncoef, ccoef, natoms);

  edge_scatter_kernel<<<(nedges + NEDGE_BLK - 1) / NEDGE_BLK, NEDGE_BLK, 0, stream>>>(
      cart, nl, shifts, neigh_factor, ncoef, co_a, nedges);

  contract_density_kernel<<<(natoms * 64 + 255) / 256, 256, 0, stream>>>(
      co_a, cc, ccoef, density, natoms);

  float* co_cur = co_a;
  float* co_nxt = co_b;
  const int mlp_grid = (natoms + 63) / 64;

  for (int t = 0; t < 3; ++t) {
    const int b = 19 + 8 * t;
    mlp_wmma_kernel<<<mlp_grid, 128, 0, stream>>>(density,
        (const float*)d_in[b + 0], (const float*)d_in[b + 1],
        (const float*)d_in[b + 2], (const float*)d_in[b + 3],
        (const float*)d_in[b + 4], (const float*)d_in[b + 5],
        (const float*)d_in[b + 6], (const float*)d_in[b + 7],
        8, icoef, nullptr, natoms);

    hipMemcpyAsync(co_nxt, co_cur, (size_t)natoms * 72 * sizeof(float),
                   hipMemcpyDeviceToDevice, stream);

    iter_edge_kernel<<<(nedges + NEDGE_BLK - 1) / NEDGE_BLK, NEDGE_BLK, 0, stream>>>(
        cart, nl, shifts, neigh_factor, ncoef, icoef, co_cur, co_nxt, nedges);

    { float* tmp = co_cur; co_cur = co_nxt; co_nxt = tmp; }

    contract_density_kernel<<<(natoms * 64 + 255) / 256, 256, 0, stream>>>(
        co_cur, cc + (size_t)(t + 1) * 3 * 8 * 64, ccoef, density, natoms);
  }

  hipMemsetAsync(d_out, 0, sizeof(float), stream);
  mlp_wmma_kernel<<<mlp_grid, 128, 0, stream>>>(density,
      (const float*)d_in[43], (const float*)d_in[44],
      (const float*)d_in[45], (const float*)d_in[46],
      (const float*)d_in[47], (const float*)d_in[48],
      (const float*)d_in[49], (const float*)d_in[50],
      1, (float*)d_out, center_factor, natoms);
}
